// CrossAttentionFusion_72739566125535
// MI455X (gfx1250) — matrix operand order repaired and checked by reading, not compiled, not run on hardware
//
#include <hip/hip_runtime.h>

typedef __attribute__((ext_vector_type(16))) _Float16 v16h;
typedef __attribute__((ext_vector_type(8)))  float    v8f;

#define FD     512
#define GDIM   1024
#define CDIM   256
#define LTOK   4096
#define BATCH  32
#define TM     16
#define NBLK   (LTOK / TM)   /* 256 token tiles per batch */
#define NHEAD  8
#define DH     64
#define ATT_SCALE 0.125f     /* 1/sqrt(64) */
#define LN_EPS 1e-5f
#define BN_EPS 1e-5f

/* LDS strides padded to avoid 64-bank conflicts on row-strided fragment loads */
#define ASTR   264   /* f16 elems per row of A tile (256 + 8)  */
#define XSTR   513   /* f32 elems per row of x/v tile (512 + 1) */
#define XBSTR  520   /* f16 elems per row of x-f16 tile (512+8) */

__device__ __forceinline__ void block_reduce2(float& s1, float& s2, float* red) {
  #pragma unroll
  for (int o = 16; o > 0; o >>= 1) {
    s1 += __shfl_xor(s1, o, 32);
    s2 += __shfl_xor(s2, o, 32);
  }
  const int lane = threadIdx.x & 31;
  const int wv   = threadIdx.x >> 5;
  if (lane == 0) { red[wv] = s1; red[8 + wv] = s2; }
  __syncthreads();
  float a = 0.f, c = 0.f;
  #pragma unroll
  for (int i = 0; i < 8; ++i) { a += red[i]; c += red[8 + i]; }
  s1 = a; s2 = c;
  __syncthreads();
}

/* -------- Kernel P: one-time f32 -> f16 weight conversion (L2-resident) -- */
__global__ __launch_bounds__(256)
void convert_weights(const float* __restrict__ Wl, const float* __restrict__ Wk,
                     const float* __restrict__ Wv,
                     _Float16* __restrict__ Wl16, _Float16* __restrict__ Wk16,
                     _Float16* __restrict__ Wv16) {
  const int i = blockIdx.x * 256 + threadIdx.x;   /* grid covers 512*512 */
  if (i < FD * CDIM) Wl16[i] = (_Float16)Wl[i];
  if (i < FD * FD) {
    Wk16[i] = (_Float16)Wk[i];
    Wv16[i] = (_Float16)Wv[i];
  }
}

/* ---------------- Kernel A: global branch (g) and query projection (q) --- */
__global__ __launch_bounds__(256)
void global_branch(const float* __restrict__ gfeat,
                   const float* __restrict__ Wg, const float* __restrict__ bg,
                   const float* __restrict__ gn_g, const float* __restrict__ gn_b,
                   const float* __restrict__ Wq, const float* __restrict__ bq,
                   float* __restrict__ ws_g, float* __restrict__ ws_q) {
  __shared__ float gin[GDIM];
  __shared__ float gl[FD];
  __shared__ float red[16];
  const int tid = threadIdx.x;
  const int b   = blockIdx.x;

  for (int i = tid; i < GDIM; i += 256) gin[i] = gfeat[(size_t)b * GDIM + i];
  __syncthreads();

  float t[2]; float s1 = 0.f, s2 = 0.f;
  #pragma unroll
  for (int k = 0; k < 2; ++k) {
    const int f = tid + k * 256;
    float z = bg[f];
    const float* wr = Wg + (size_t)f * GDIM;
    for (int j = 0; j < GDIM; ++j) z += gin[j] * wr[j];
    t[k] = z; s1 += z; s2 += z * z;
  }
  block_reduce2(s1, s2, red);
  const float mean = s1 * (1.f / FD);
  const float rstd = rsqrtf(s2 * (1.f / FD) - mean * mean + LN_EPS);
  #pragma unroll
  for (int k = 0; k < 2; ++k) {
    const int f = tid + k * 256;
    const float y = (t[k] - mean) * rstd * gn_g[f] + gn_b[f];
    gl[f] = y;
    ws_g[(size_t)b * FD + f] = y;
  }
  __syncthreads();
  #pragma unroll
  for (int k = 0; k < 2; ++k) {
    const int f = tid + k * 256;
    float z = bq[f];
    const float* wr = Wq + (size_t)f * FD;
    for (int j = 0; j < FD; ++j) z += gl[j] * wr[j];
    ws_q[(size_t)b * FD + f] = z;
  }
}

/* -------- Kernel B: fused conv1x1 + BN + ReLU + LN + K/V proj + partial
 *          flash-attention per 16-token tile, all via v_wmma f16 ----------- */
__global__ __launch_bounds__(256)
void fused_local_attn(const float* __restrict__ local,
                      const _Float16* __restrict__ Wl16,
                      const float* __restrict__ bn_g, const float* __restrict__ bn_b,
                      const float* __restrict__ bn_m, const float* __restrict__ bn_v,
                      const float* __restrict__ ln_g, const float* __restrict__ ln_b,
                      const _Float16* __restrict__ Wk16, const float* __restrict__ bk,
                      const _Float16* __restrict__ Wv16, const float* __restrict__ bv,
                      const float* __restrict__ q_all,
                      float* __restrict__ ws_ml, float* __restrict__ ws_acc) {
  __shared__ _Float16 Albuf[TM * ASTR];      /* local tile, f16 (M=16,K=256)   */
  __shared__ float    xf[TM * XSTR];         /* x f32, later reused as v tile  */
  __shared__ _Float16 xbuf[TM * XBSTR];      /* normalized x, f16 (M=16,K=512) */
  __shared__ float    sbuf[TM * NHEAD];      /* score accumulators             */
  __shared__ float    pbuf[TM * NHEAD];      /* softmax numerators             */
  __shared__ float    mls[NHEAD];

  const int tid  = threadIdx.x;
  const int b    = blockIdx.x / NBLK;
  const int blk  = blockIdx.x % NBLK;
  const int t0   = blk * TM;
  const int lane = tid & 31;
  const int wv   = tid >> 5;
  const int half = lane >> 4;
  const int nn   = lane & 15;    /* A row m / B,C column n */

  if (tid < TM * NHEAD) sbuf[tid] = 0.f;

  /* load 16-token x 256-channel tile; local layout is (B, C, L) */
  const float* lptr = local + ((size_t)b * CDIM) * LTOK + t0;
  for (int idx = tid; idx < TM * CDIM; idx += 256) {
    const int m = idx & (TM - 1);
    const int c = idx >> 4;
    Albuf[m * ASTR + c] = (_Float16)lptr[(size_t)c * LTOK + m];
  }
  __syncthreads();

  /* ---- GEMM1: x(16x512) = A(16x256) * Wl^T ---- */
  for (int nt = 0; nt < 4; ++nt) {
    const int n0 = (wv * 4 + nt) * 16;
    v8f acc = {0.f, 0.f, 0.f, 0.f, 0.f, 0.f, 0.f, 0.f};
    for (int ks = 0; ks < CDIM / 32; ++ks) {
      const int kk = ks * 32;
      v16h afr;
      #pragma unroll
      for (int j = 0; j < 16; ++j) {
        const int K = j + (half << 4);
        afr[j] = Albuf[nn * ASTR + kk + K];
      }
      /* B fragment = 16 contiguous f16, 32B-aligned -> 2x global_load_b128 */
      const v16h bfr = *(const v16h*)(Wl16 + (size_t)(n0 + nn) * CDIM + kk + half * 16);
      acc = __builtin_amdgcn_wmma_f32_16x16x32_f16(false, afr, false, bfr,
                                                   (short)0, acc, false, false);
    }
    #pragma unroll
    for (int r = 0; r < 8; ++r) xf[(r + half * 8) * XSTR + n0 + nn] = acc[r];
  }
  __syncthreads();

  /* ---- BN(eval) + ReLU + LayerNorm per token -> f16 ---- */
  #pragma unroll
  for (int tk = 0; tk < 2; ++tk) {
    const int m = wv + tk * 8;
    float vals[16];
    float s1 = 0.f, s2 = 0.f;
    #pragma unroll
    for (int i = 0; i < 16; ++i) {
      const int f = lane + 32 * i;
      float t = xf[m * XSTR + f];
      t = (t - bn_m[f]) * rsqrtf(bn_v[f] + BN_EPS) * bn_g[f] + bn_b[f];
      t = fmaxf(t, 0.f);
      vals[i] = t; s1 += t; s2 += t * t;
    }
    #pragma unroll
    for (int o = 16; o > 0; o >>= 1) {
      s1 += __shfl_xor(s1, o, 32);
      s2 += __shfl_xor(s2, o, 32);
    }
    const float mean = s1 * (1.f / FD);
    const float rstd = rsqrtf(s2 * (1.f / FD) - mean * mean + LN_EPS);
    #pragma unroll
    for (int i = 0; i < 16; ++i) {
      const int f = lane + 32 * i;
      xbuf[m * XBSTR + f] = (_Float16)((vals[i] - mean) * rstd * ln_g[f] + ln_b[f]);
    }
  }
  __syncthreads();

  /* ---- merged GEMM2+3: K and V projection share the A fragment.
   *      Two independent WMMA chains per k-step (no D->A/B hazard). ---- */
  const float* qrow = q_all + (size_t)b * FD;
  for (int nt = 0; nt < 4; ++nt) {
    const int n0 = (wv * 4 + nt) * 16;
    const int h  = n0 >> 6;
    v8f accK = {0.f, 0.f, 0.f, 0.f, 0.f, 0.f, 0.f, 0.f};
    v8f accV = {0.f, 0.f, 0.f, 0.f, 0.f, 0.f, 0.f, 0.f};
    for (int ks = 0; ks < FD / 32; ++ks) {
      const int kk = ks * 32;
      v16h afr;
      #pragma unroll
      for (int j = 0; j < 16; ++j) {
        const int K = j + (half << 4);
        afr[j] = xbuf[nn * XBSTR + kk + K];
      }
      const size_t woff = (size_t)(n0 + nn) * FD + kk + half * 16;
      const v16h bK = *(const v16h*)(Wk16 + woff);
      const v16h bV = *(const v16h*)(Wv16 + woff);
      accK = __builtin_amdgcn_wmma_f32_16x16x32_f16(false, afr, false, bK,
                                                    (short)0, accK, false, false);
      accV = __builtin_amdgcn_wmma_f32_16x16x32_f16(false, afr, false, bV,
                                                    (short)0, accV, false, false);
    }
    /* K epilogue: fold q.k into per-token/head score partials */
    const float qv  = qrow[n0 + nn];
    const float bkk = bk[n0 + nn];
    float part[8];
    #pragma unroll
    for (int r = 0; r < 8; ++r) part[r] = (accK[r] + bkk) * qv;
    #pragma unroll
    for (int o = 1; o < 16; o <<= 1) {
      #pragma unroll
      for (int r = 0; r < 8; ++r) part[r] += __shfl_xor(part[r], o, 32);
    }
    if (nn == 0) {
      #pragma unroll
      for (int r = 0; r < 8; ++r)
        atomicAdd(&sbuf[(r + half * 8) * NHEAD + h], part[r] * ATT_SCALE);
    }
    /* V epilogue: v tile -> LDS (reuses xf) */
    const float bvv = bv[n0 + nn];
    #pragma unroll
    for (int r = 0; r < 8; ++r) xf[(r + half * 8) * XSTR + n0 + nn] = accV[r] + bvv;
  }
  __syncthreads();

  /* ---- per-tile online softmax partials ---- */
  if (tid < NHEAD) {
    float mx = -1e30f;
    for (int m = 0; m < TM; ++m) mx = fmaxf(mx, sbuf[m * NHEAD + tid]);
    mls[tid] = mx;
  }
  __syncthreads();
  if (tid < TM * NHEAD) pbuf[tid] = __expf(sbuf[tid] - mls[tid & 7]);
  __syncthreads();
  if (tid < NHEAD) {
    float ls = 0.f;
    for (int m = 0; m < TM; ++m) ls += pbuf[m * NHEAD + tid];
    const size_t idx = ((size_t)b * NHEAD + tid) * NBLK + blk;
    ws_ml[idx * 2]     = mls[tid];
    ws_ml[idx * 2 + 1] = ls;
  }
  __syncthreads();
  for (int n = tid; n < FD; n += 256) {
    const int h = n >> 6;
    float a = 0.f;
    #pragma unroll
    for (int m = 0; m < TM; ++m) a += pbuf[m * NHEAD + h] * xf[m * XSTR + n];
    ws_acc[(((size_t)b * NHEAD + h) * NBLK + blk) * DH + (n & 63)] = a;
  }
}

/* -------- Kernel C: flash combine + Wo + gate fusion + GELU MLP + LN ----- */
__global__ __launch_bounds__(256)
void combine_tail(const float* __restrict__ ws_g,
                  const float* __restrict__ ws_ml, const float* __restrict__ ws_acc,
                  const float* __restrict__ Wo, const float* __restrict__ bo,
                  const float* __restrict__ Wgate, const float* __restrict__ bgate,
                  const float* __restrict__ We1, const float* __restrict__ be1,
                  const float* __restrict__ We2, const float* __restrict__ be2,
                  const float* __restrict__ eg, const float* __restrict__ eb,
                  float* __restrict__ out) {
  __shared__ float wbuf[NBLK * NHEAD];
  __shared__ float gM[NHEAD], Ls[NHEAD];
  __shared__ float olds[FD], glds[FD], attb[FD], fusb[FD], h1b[2 * FD], red[16];
  const int tid = threadIdx.x;
  const int b   = blockIdx.x;

  if (tid < NHEAD) {
    float mx = -1e30f;
    const size_t base = ((size_t)b * NHEAD + tid) * NBLK;
    for (int k = 0; k < NBLK; ++k) mx = fmaxf(mx, ws_ml[(base + k) * 2]);
    gM[tid] = mx;
  }
  __syncthreads();
  {
    const int blk = tid;  /* blockDim == NBLK == 256 */
    #pragma unroll
    for (int h = 0; h < NHEAD; ++h) {
      const size_t idx = ((size_t)b * NHEAD + h) * NBLK + blk;
      wbuf[blk * NHEAD + h] = __expf(ws_ml[idx * 2] - gM[h]);
    }
  }
  __syncthreads();
  if (tid < NHEAD) {
    float L = 0.f;
    const size_t base = ((size_t)b * NHEAD + tid) * NBLK;
    for (int k = 0; k < NBLK; ++k) L += ws_ml[(base + k) * 2 + 1] * wbuf[k * NHEAD + tid];
    Ls[tid] = L;
  }
  __syncthreads();
  for (int n = tid; n < FD; n += 256) {
    const int h = n >> 6;
    const size_t base = (((size_t)b * NHEAD + h) * NBLK) * DH + (n & 63);
    float a = 0.f;
    for (int k = 0; k < NBLK; ++k) a += ws_acc[base + (size_t)k * DH] * wbuf[k * NHEAD + h];
    olds[n] = a / Ls[h];
    glds[n] = ws_g[(size_t)b * FD + n];
  }
  __syncthreads();

  /* attended = o @ Wo^T + bo */
  #pragma unroll
  for (int k = 0; k < 2; ++k) {
    const int f = tid + k * 256;
    float z = bo[f];
    const float* wr = Wo + (size_t)f * FD;
    for (int j = 0; j < FD; ++j) z += olds[j] * wr[j];
    attb[f] = z;
  }
  __syncthreads();

  /* gate = sigmoid([g, att] @ Wgate^T + bgate); fused = gate*g + (1-gate)*att */
  #pragma unroll
  for (int k = 0; k < 2; ++k) {
    const int f = tid + k * 256;
    float z = bgate[f];
    const float* wr = Wgate + (size_t)f * (2 * FD);
    for (int j = 0; j < FD; ++j) z += glds[j] * wr[j];
    for (int j = 0; j < FD; ++j) z += attb[j] * wr[FD + j];
    const float gte = 1.f / (1.f + __expf(-z));
    fusb[f] = gte * glds[f] + (1.f - gte) * attb[f];
  }
  __syncthreads();

  /* h1 = gelu_exact(fused @ We1^T + be1) */
  #pragma unroll
  for (int k = 0; k < 4; ++k) {
    const int e = tid + k * 256;
    float z = be1[e];
    const float* wr = We1 + (size_t)e * FD;
    for (int j = 0; j < FD; ++j) z += fusb[j] * wr[j];
    h1b[e] = 0.5f * z * (1.f + erff(z * 0.70710678118654752f));
  }
  __syncthreads();

  /* out = LN(h1 @ We2^T + be2) */
  float t[2]; float s1 = 0.f, s2 = 0.f;
  #pragma unroll
  for (int k = 0; k < 2; ++k) {
    const int f = tid + k * 256;
    float z = be2[f];
    const float* wr = We2 + (size_t)f * (2 * FD);
    for (int j = 0; j < 2 * FD; ++j) z += h1b[j] * wr[j];
    t[k] = z; s1 += z; s2 += z * z;
  }
  block_reduce2(s1, s2, red);
  const float mean = s1 * (1.f / FD);
  const float rstd = rsqrtf(s2 * (1.f / FD) - mean * mean + LN_EPS);
  #pragma unroll
  for (int k = 0; k < 2; ++k) {
    const int f = tid + k * 256;
    out[(size_t)b * FD + f] = (t[k] - mean) * rstd * eg[f] + eb[f];
  }
}

extern "C" void kernel_launch(void* const* d_in, const int* in_sizes, int n_in,
                              void* d_out, int out_size, void* d_ws, size_t ws_size,
                              hipStream_t stream) {
  (void)in_sizes; (void)n_in; (void)out_size; (void)ws_size;
  const float* gfeat = (const float*)d_in[0];
  const float* local = (const float*)d_in[1];
  const float* Wg    = (const float*)d_in[2];
  const float* bg    = (const float*)d_in[3];
  const float* gn_g  = (const float*)d_in[4];
  const float* gn_b  = (const float*)d_in[5];
  const float* Wl    = (const float*)d_in[6];
  const float* bn_g  = (const float*)d_in[7];
  const float* bn_b  = (const float*)d_in[8];
  const float* bn_m  = (const float*)d_in[9];
  const float* bn_v  = (const float*)d_in[10];
  const float* ln_g  = (const float*)d_in[11];
  const float* ln_b  = (const float*)d_in[12];
  const float* Wq    = (const float*)d_in[13];
  const float* bq    = (const float*)d_in[14];
  const float* Wk    = (const float*)d_in[15];
  const float* bk    = (const float*)d_in[16];
  const float* Wv    = (const float*)d_in[17];
  const float* bv    = (const float*)d_in[18];
  const float* Wo    = (const float*)d_in[19];
  const float* bo    = (const float*)d_in[20];
  const float* Wgate = (const float*)d_in[21];
  const float* bgate = (const float*)d_in[22];
  const float* We1   = (const float*)d_in[23];
  const float* be1   = (const float*)d_in[24];
  const float* We2   = (const float*)d_in[25];
  const float* be2   = (const float*)d_in[26];
  const float* eg    = (const float*)d_in[27];
  const float* eb    = (const float*)d_in[28];

  float* wsf    = (float*)d_ws;
  float* ws_g   = wsf;                                      /* 32*512         */
  float* ws_q   = ws_g + (size_t)BATCH * FD;                /* 32*512         */
  float* ws_ml  = ws_q + (size_t)BATCH * FD;                /* 32*8*256*2     */
  float* ws_acc = ws_ml + (size_t)BATCH * NHEAD * NBLK * 2; /* 32*8*256*64    */
  /* f16 weight cache, 32B-aligned (float region is a multiple of 32B) */
  _Float16* wl16 = (_Float16*)(ws_acc + (size_t)BATCH * NHEAD * NBLK * DH);
  _Float16* wk16 = wl16 + (size_t)FD * CDIM;
  _Float16* wv16 = wk16 + (size_t)FD * FD;

  convert_weights<<<(FD * FD) / 256, 256, 0, stream>>>(Wl, Wk, Wv, wl16, wk16, wv16);
  global_branch<<<BATCH, 256, 0, stream>>>(gfeat, Wg, bg, gn_g, gn_b, Wq, bq,
                                           ws_g, ws_q);
  fused_local_attn<<<BATCH * NBLK, 256, 0, stream>>>(local, wl16, bn_g, bn_b, bn_m,
                                                     bn_v, ln_g, ln_b, wk16, bk, wv16,
                                                     bv, ws_q, ws_ml, ws_acc);
  combine_tail<<<BATCH, 256, 0, stream>>>(ws_g, ws_ml, ws_acc, Wo, bo, Wgate,
                                          bgate, We1, be1, We2, be2, eg, eb,
                                          (float*)d_out);
}
